// MaskDecoder2_38946763441078
// MI455X (gfx1250) — compile-verified, matrix operand order checked by reading, not run on hardware
//
#include <hip/hip_runtime.h>
#include <hip/hip_bf16.h>
#include <math.h>

// Problem constants (reference: B,N,D,H = 2,2048,1024,8)
#define B_   2
#define N_   2048
#define D_   1024
#define H_   8
#define DH_  128
#define TDH_ 256      // 2*DH
#define D2_  2048     // 2*D
#define LAMBDA_INIT_ 0.8f
#define EPS_ 1e-5f

#define USE_ASYNC 1   // gfx1250 GLOBAL_LOAD_ASYNC_TO_LDS_B128 staging

typedef __bf16 bf16_t;
typedef __attribute__((ext_vector_type(16))) __bf16 v16bf;
typedef __attribute__((ext_vector_type(8)))  float  v8f;

static __device__ __forceinline__ v8f wmma_bf16f32(v16bf a, v16bf b, v8f c) {
  // D = A(16x32 bf16) * B(32x16 bf16) + C(16x16 f32)
  return __builtin_amdgcn_wmma_f32_16x16x32_bf16(
      /*neg_a=*/false, a, /*neg_b=*/false, b,
      /*c_mod=*/(short)0, c, /*reuse_a=*/false, /*reuse_b=*/false);
}

// 16-bit matrix fragment: lanes 0-15 hold K {0..7,16..23}; lanes 16-31 hold
// K {8..15,24..31}.  With data contiguous along K per lane, a fragment is two
// ds_load_b128 at (base + hi*16B) and (base + hi*16B + 32B).
static __device__ __forceinline__ v16bf frag_from_contig(const bf16_t* base_k0, int hi) {
  v16bf f;
  const bf16_t* p0 = base_k0 + hi * 8;        // elements 0..7  -> K hi*8 + 0..7
  const bf16_t* p1 = base_k0 + 16 + hi * 8;   // elements 8..15 -> K hi*8 + 16..23
  #pragma unroll
  for (int e = 0; e < 8; ++e) { f[e] = p0[e]; f[e + 8] = p1[e]; }
  return f;
}

// Async DMA: global -> LDS, 16B per lane, tracked by ASYNCcnt (no VGPR data).
static __device__ __forceinline__ void async_ld_b128(bf16_t* lds_dst, const bf16_t* gsrc) {
#if USE_ASYNC
  unsigned lds_off = (unsigned)(uintptr_t)lds_dst;           // LDS aperture: addr[31:0]
  unsigned long long ga = (unsigned long long)(uintptr_t)gsrc;
  asm volatile("global_load_async_to_lds_b128 %0, %1, off" :: "v"(lds_off), "v"(ga) : "memory");
#else
  *(uint4*)lds_dst = *(const uint4*)gsrc;
#endif
}
static __device__ __forceinline__ void async_wait_all() {
#if USE_ASYNC
  asm volatile("s_wait_asynccnt 0x0" ::: "memory");
#endif
}

// ---------------------------------------------------------------------------
// fp32 -> bf16 conversion (no transpose) for X
// ---------------------------------------------------------------------------
__global__ void cvt_f32_bf16(const float* __restrict__ in, bf16_t* __restrict__ out, int n) {
  int i = blockIdx.x * 256 + threadIdx.x;
  if (i < n) out[i] = (bf16_t)in[i];
}

// ---------------------------------------------------------------------------
// fp32 [R x C] -> bf16 transposed [C x R], 32x32 LDS tiles (dims % 32 == 0)
// ---------------------------------------------------------------------------
__global__ __launch_bounds__(256)
void cvt_transpose_bf16(const float* __restrict__ in, bf16_t* __restrict__ out, int R, int C) {
  __shared__ float tile[32][33];
  const int c0 = blockIdx.x * 32, r0 = blockIdx.y * 32;
  const int tx = threadIdx.x & 31, ty = threadIdx.x >> 5;   // ty: 0..7
  #pragma unroll
  for (int i = 0; i < 32; i += 8)
    tile[ty + i][tx] = in[(size_t)(r0 + ty + i) * C + c0 + tx];
  __syncthreads();
  #pragma unroll
  for (int i = 0; i < 32; i += 8)
    out[(size_t)(c0 + ty + i) * R + r0 + tx] = (bf16_t)tile[tx][ty + i];
}

// ---------------------------------------------------------------------------
// lambda[h] = exp(<lq1,lk1>) - exp(<lq2,lk2>) + lambda_init   (one wave/head)
// ---------------------------------------------------------------------------
__global__ void lambda_kernel(const float* __restrict__ lq1, const float* __restrict__ lk1,
                              const float* __restrict__ lq2, const float* __restrict__ lk2,
                              float* __restrict__ lam) {
  int w = threadIdx.x >> 5, lane = threadIdx.x & 31;
  if (w < H_) {
    float s1 = 0.f, s2 = 0.f;
    for (int i = lane; i < DH_; i += 32) {
      s1 += lq1[w * DH_ + i] * lk1[w * DH_ + i];
      s2 += lq2[w * DH_ + i] * lk2[w * DH_ + i];
    }
    #pragma unroll
    for (int off = 16; off; off >>= 1) {
      s1 += __shfl_xor(s1, off, 32);
      s2 += __shfl_xor(s2, off, 32);
    }
    if (lane == 0) lam[w] = expf(s1) - expf(s2) + LAMBDA_INIT_;
  }
}

// ---------------------------------------------------------------------------
// QKV projection: X[4096x1024] * W -> Q/K in [B,H,N,256], V in [B,H,256,N].
// Weights pre-transposed: Wt[n][k].  128x128 tile, BK=32, 8 waves (4x2),
// double-buffered async LDS staging (one barrier per k-step).
// ---------------------------------------------------------------------------
__global__ __launch_bounds__(256, 1)
void qkv_proj_kernel(const bf16_t* __restrict__ X,
                     const bf16_t* __restrict__ Wqt, const bf16_t* __restrict__ Wkt,
                     const bf16_t* __restrict__ Wvt,
                     bf16_t* __restrict__ Qo, bf16_t* __restrict__ Ko, bf16_t* __restrict__ Vt) {
  const bf16_t* Wt; bf16_t* O; bool vmode = false;
  if (blockIdx.z == 0)      { Wt = Wqt; O = Qo; }
  else if (blockIdx.z == 1) { Wt = Wkt; O = Ko; }
  else                      { Wt = Wvt; O = Vt; vmode = true; }

  __shared__ __align__(16) bf16_t sA[2][128 * 32];   // [buf][m][k]
  __shared__ __align__(16) bf16_t sBt[2][128 * 32];  // [buf][n][k]

  const int m0 = blockIdx.y * 128;
  const int n0 = blockIdx.x * 128;
  const int tid = threadIdx.x;
  const int wave = tid >> 5, lane = tid & 31;
  const int lrow = lane & 15, hi = lane >> 4;
  const int wm = wave >> 1;  // 0..3 (32 rows each)
  const int wn = wave & 1;   // 0..1 (64 cols each)

  // per-thread staging slice: 2 x 16B for A, 2 x 16B for Bt
  const int sr0 = tid >> 2, sc0 = (tid & 3) * 8;          // idx = tid
  const int sr1 = (tid + 256) >> 2, sc1 = ((tid + 256) & 3) * 8;

  auto stage = [&](int buf, int k0) {
    async_ld_b128(&sA[buf][sr0 * 32 + sc0], X  + (size_t)(m0 + sr0) * D_ + k0 + sc0);
    async_ld_b128(&sA[buf][sr1 * 32 + sc1], X  + (size_t)(m0 + sr1) * D_ + k0 + sc1);
    async_ld_b128(&sBt[buf][sr0 * 32 + sc0], Wt + (size_t)(n0 + sr0) * D_ + k0 + sc0);
    async_ld_b128(&sBt[buf][sr1 * 32 + sc1], Wt + (size_t)(n0 + sr1) * D_ + k0 + sc1);
  };

  v8f acc[2][4] = {};
  stage(0, 0);
  int cur = 0;
  for (int k0 = 0; k0 < D_; k0 += 32) {
    async_wait_all();
    __syncthreads();
    if (k0 + 32 < D_) stage(cur ^ 1, k0 + 32);

    v16bf afrag[2], bfrag[4];
    #pragma unroll
    for (int tm = 0; tm < 2; ++tm)
      afrag[tm] = frag_from_contig(&sA[cur][(wm * 32 + tm * 16 + lrow) * 32], hi);
    #pragma unroll
    for (int tn = 0; tn < 4; ++tn)
      bfrag[tn] = frag_from_contig(&sBt[cur][(wn * 64 + tn * 16 + lrow) * 32], hi);
    #pragma unroll
    for (int tm = 0; tm < 2; ++tm)
      #pragma unroll
      for (int tn = 0; tn < 4; ++tn)
        acc[tm][tn] = wmma_bf16f32(afrag[tm], bfrag[tn], acc[tm][tn]);
    cur ^= 1;
  }

  // Epilogue: permute to [B,H,N,256] (Q,K) or [B,H,256,N] (V).
  if (vmode) {
    #pragma unroll
    for (int tm = 0; tm < 2; ++tm)
      #pragma unroll
      for (int tn = 0; tn < 4; ++tn)
        #pragma unroll
        for (int r = 0; r < 8; ++r) {
          int row = m0 + wm * 32 + tm * 16 + r + hi * 8;
          int col = n0 + wn * 64 + tn * 16 + lrow;
          int b = row / N_, n = row % N_;
          int h = col / TDH_, j = col % TDH_;
          O[(((size_t)(b * H_ + h)) * TDH_ + j) * N_ + n] = (bf16_t)acc[tm][tn][r];
        }
  } else {
    #pragma unroll
    for (int tm = 0; tm < 2; ++tm)
      #pragma unroll
      for (int tn = 0; tn < 4; ++tn)
        #pragma unroll
        for (int r = 0; r < 8; ++r) {
          int row = m0 + wm * 32 + tm * 16 + r + hi * 8;
          int col = n0 + wn * 64 + tn * 16 + lrow;
          int b = row / N_, n = row % N_;
          int h = col / TDH_, j = col % TDH_;
          O[(((size_t)(b * H_ + h)) * N_ + n) * TDH_ + j] = (bf16_t)acc[tm][tn][r];
        }
  }
}

// ---------------------------------------------------------------------------
// Differential flash attention, double-buffered async K/V staging.
// grid: (N/64, B*H).  8 waves: (wq 0..3) x 16 q-rows, (wc 0..1) x 128 V-cols.
// ---------------------------------------------------------------------------
#define KV_BYTES (64*256*2 + 256*64*2)   // sK + sVt per buffer (64 KB)
#define SMEM_ATTN (2*KV_BYTES + 8*2*16*64*2 /*sP*/ + 64*2*4 /*stats*/)

__global__ __launch_bounds__(256, 1)
void diff_attn_kernel(const bf16_t* __restrict__ Q, const bf16_t* __restrict__ Kc,
                      const bf16_t* __restrict__ Vt, const float* __restrict__ lamArr,
                      const float* __restrict__ rms_scale, bf16_t* __restrict__ AO) {
  extern __shared__ char smem_raw[];
  bf16_t* sKbuf[2];  bf16_t* sVbuf[2];
  sKbuf[0] = (bf16_t*)smem_raw;                    // [64][256]
  sVbuf[0] = sKbuf[0] + 64 * 256;                  // [256][64]
  sKbuf[1] = sVbuf[0] + 256 * 64;
  sVbuf[1] = sKbuf[1] + 64 * 256;
  bf16_t* sP = sVbuf[1] + 256 * 64;                // [8 waves][2][16][64]
  float*  sStat = (float*)(sP + 8 * 2 * 16 * 64);  // [64][2]

  const int qb = blockIdx.x;
  const int bh = blockIdx.y;
  const int h = bh % H_, b = bh / H_;
  const int tid = threadIdx.x, wave = tid >> 5, lane = tid & 31;
  const int lrow = lane & 15, hi = lane >> 4;
  const int wq = wave >> 1, wc = wave & 1;
  const float lam = lamArr[h];
  const float scale = 0.08838834764831845f;  // 1/sqrt(128)
  const size_t baseRow = (size_t)bh * N_ * TDH_;   // Q/K row-major base
  const size_t baseCol = (size_t)bh * TDH_ * N_;   // Vt base (rows j, stride N_)
  const int q0 = qb * 64;

  auto stage = [&](int buf, int kb) {
    const int k0 = kb * 64;
    const bf16_t* Kg = Kc + baseRow + (size_t)k0 * TDH_;
    const bf16_t* Vg = Vt + baseCol + k0;
    #pragma unroll
    for (int p = 0; p < 8; ++p) {                 // sK: 64x256
      int idx = tid + p * 256;
      int r = idx >> 5, c = (idx & 31) * 8;
      async_ld_b128(&sKbuf[buf][r * 256 + c], &Kg[(size_t)r * TDH_ + c]);
    }
    #pragma unroll
    for (int p = 0; p < 8; ++p) {                 // sVt: 256x64
      int idx = tid + p * 256;
      int j = idx >> 3, c = (idx & 7) * 8;
      async_ld_b128(&sVbuf[buf][j * 64 + c], &Vg[(size_t)j * N_ + c]);
    }
  };

  // Resident Q fragments (global rows contiguous -> global_load_b128).
  v16bf q1f[4], q2f[4];
  {
    const bf16_t* qrow = Q + baseRow + (size_t)(q0 + wq * 16 + lrow) * TDH_;
    #pragma unroll
    for (int kc = 0; kc < 4; ++kc) {
      q1f[kc] = frag_from_contig(qrow + kc * 32, hi);
      q2f[kc] = frag_from_contig(qrow + 128 + kc * 32, hi);
    }
  }

  v8f o1[8] = {}, o2[8] = {};
  float m1v[8], m2v[8], l1v[8], l2v[8];
  #pragma unroll
  for (int r = 0; r < 8; ++r) { m1v[r] = -3.0e38f; m2v[r] = -3.0e38f; l1v[r] = 0.f; l2v[r] = 0.f; }

  bf16_t* myP1 = sP + (wave * 2 + 0) * 16 * 64;
  bf16_t* myP2 = sP + (wave * 2 + 1) * 16 * 64;

  stage(0, 0);
  int cur = 0;
  for (int kb = 0; kb <= qb; ++kb) {
    const int k0 = kb * 64;
    async_wait_all();
    __syncthreads();
    if (kb + 1 <= qb) stage(cur ^ 1, kb + 1);
    bf16_t* sK  = sKbuf[cur];
    bf16_t* sVc = sVbuf[cur];

    // S1 = Q1*K1^T, S2 = Q2*K2^T : 16 x 64 (4 n-tiles); K rows contiguous.
    v8f s1[4] = {}, s2[4] = {};
    #pragma unroll
    for (int tn = 0; tn < 4; ++tn) {
      const bf16_t* krow = &sK[(tn * 16 + lrow) * 256];
      #pragma unroll
      for (int kc = 0; kc < 4; ++kc) {
        v16bf b1 = frag_from_contig(krow + kc * 32, hi);
        v16bf b2 = frag_from_contig(krow + 128 + kc * 32, hi);
        s1[tn] = wmma_bf16f32(q1f[kc], b1, s1[tn]);
        s2[tn] = wmma_bf16f32(q2f[kc], b2, s2[tn]);
      }
    }

    // scale + causal mask
    #pragma unroll
    for (int tn = 0; tn < 4; ++tn)
      #pragma unroll
      for (int r = 0; r < 8; ++r) {
        int qrow = q0 + wq * 16 + r + hi * 8;
        int kcol = k0 + tn * 16 + lrow;
        float v1 = s1[tn][r] * scale, v2 = s2[tn][r] * scale;
        if (kcol > qrow) { v1 = -3.0e38f; v2 = -3.0e38f; }
        s1[tn][r] = v1; s2[tn][r] = v2;
      }

    // dual online softmax per row (r + hi*8); P tiles to per-wave LDS
    #pragma unroll
    for (int r = 0; r < 8; ++r) {
      float rm1 = fmaxf(fmaxf(s1[0][r], s1[1][r]), fmaxf(s1[2][r], s1[3][r]));
      float rm2 = fmaxf(fmaxf(s2[0][r], s2[1][r]), fmaxf(s2[2][r], s2[3][r]));
      #pragma unroll
      for (int off = 1; off < 16; off <<= 1) {
        rm1 = fmaxf(rm1, __shfl_xor(rm1, off, 32));
        rm2 = fmaxf(rm2, __shfl_xor(rm2, off, 32));
      }
      float nm1 = fmaxf(m1v[r], rm1), nm2 = fmaxf(m2v[r], rm2);
      float c1 = expf(m1v[r] - nm1), c2 = expf(m2v[r] - nm2);
      float p1[4], p2[4], ps1 = 0.f, ps2 = 0.f;
      #pragma unroll
      for (int tn = 0; tn < 4; ++tn) {
        p1[tn] = expf(s1[tn][r] - nm1);  ps1 += p1[tn];
        p2[tn] = expf(s2[tn][r] - nm2);  ps2 += p2[tn];
      }
      #pragma unroll
      for (int off = 1; off < 16; off <<= 1) {
        ps1 += __shfl_xor(ps1, off, 32);
        ps2 += __shfl_xor(ps2, off, 32);
      }
      l1v[r] = l1v[r] * c1 + ps1;  m1v[r] = nm1;
      l2v[r] = l2v[r] * c2 + ps2;  m2v[r] = nm2;
      #pragma unroll
      for (int ct = 0; ct < 8; ++ct) { o1[ct][r] *= c1; o2[ct][r] *= c2; }
      int row16 = r + hi * 8;
      #pragma unroll
      for (int tn = 0; tn < 4; ++tn) {
        myP1[row16 * 64 + tn * 16 + lrow] = (bf16_t)p1[tn];
        myP2[row16 * 64 + tn * 16 + lrow] = (bf16_t)p2[tn];
      }
    }

    asm volatile("s_wait_dscnt 0" ::: "memory");   // in-wave P store->load order

    // O += P * V  (8 col tiles, K=64 in 2 chunks); all fragments contiguous.
    #pragma unroll
    for (int kc = 0; kc < 2; ++kc) {
      v16bf pa1 = frag_from_contig(&myP1[lrow * 64 + kc * 32], hi);
      v16bf pa2 = frag_from_contig(&myP2[lrow * 64 + kc * 32], hi);
      #pragma unroll
      for (int ct = 0; ct < 8; ++ct) {
        int col = wc * 128 + ct * 16 + lrow;
        v16bf vb = frag_from_contig(&sVc[col * 64 + kc * 32], hi);
        o1[ct] = wmma_bf16f32(pa1, vb, o1[ct]);
        o2[ct] = wmma_bf16f32(pa2, vb, o2[ct]);
      }
    }
    cur ^= 1;
  }

  // Combine: O = O1/l1 - lam*O2/l2, RMS over 256 features (2 waves/row).
  float inv1[8], inv2[8];
  #pragma unroll
  for (int r = 0; r < 8; ++r) { inv1[r] = 1.f / l1v[r]; inv2[r] = lam / l2v[r]; }
  float ss[8];
  #pragma unroll
  for (int r = 0; r < 8; ++r) {
    float acc = 0.f;
    #pragma unroll
    for (int ct = 0; ct < 8; ++ct) {
      float v = o1[ct][r] * inv1[r] - o2[ct][r] * inv2[r];
      o1[ct][r] = v;
      acc += v * v;
    }
    #pragma unroll
    for (int off = 1; off < 16; off <<= 1) acc += __shfl_xor(acc, off, 32);
    ss[r] = acc;
  }
  __syncthreads();
  if (lrow == 0)
    #pragma unroll
    for (int r = 0; r < 8; ++r)
      sStat[(wq * 16 + r + hi * 8) * 2 + wc] = ss[r];
  __syncthreads();

  float invr[8];
  #pragma unroll
  for (int r = 0; r < 8; ++r) {
    int row16 = wq * 16 + r + hi * 8;
    float tot = sStat[row16 * 2 + 0] + sStat[row16 * 2 + 1];
    invr[r] = 1.f / sqrtf(tot * (1.0f / 256.0f) + EPS_);
  }
  const float outScale = 1.0f - LAMBDA_INIT_;
  #pragma unroll
  for (int ct = 0; ct < 8; ++ct) {
    int j = wc * 128 + ct * 16 + lrow;
    float rs = rms_scale[j] * outScale;
    #pragma unroll
    for (int r = 0; r < 8; ++r) {
      int nrow = q0 + wq * 16 + r + hi * 8;
      float val = o1[ct][r] * invr[r] * rs;
      AO[((size_t)b * N_ + nrow) * D2_ + h * TDH_ + j] = (bf16_t)val;
    }
  }
}

// ---------------------------------------------------------------------------
// Output projection: AO[4096x2048] bf16 * Wo_t[1024(n) x 2048(k)] -> f32,
// double-buffered async staging.
// ---------------------------------------------------------------------------
__global__ __launch_bounds__(256, 1)
void out_proj_kernel(const bf16_t* __restrict__ A, const bf16_t* __restrict__ Wt,
                     float* __restrict__ Cout) {
  __shared__ __align__(16) bf16_t sA[2][128 * 32];
  __shared__ __align__(16) bf16_t sBt[2][128 * 32];

  const int m0 = blockIdx.y * 128;
  const int n0 = blockIdx.x * 128;
  const int tid = threadIdx.x;
  const int wave = tid >> 5, lane = tid & 31;
  const int lrow = lane & 15, hi = lane >> 4;
  const int wm = wave >> 1, wn = wave & 1;

  const int sr0 = tid >> 2, sc0 = (tid & 3) * 8;
  const int sr1 = (tid + 256) >> 2, sc1 = ((tid + 256) & 3) * 8;

  auto stage = [&](int buf, int k0) {
    async_ld_b128(&sA[buf][sr0 * 32 + sc0], A  + (size_t)(m0 + sr0) * D2_ + k0 + sc0);
    async_ld_b128(&sA[buf][sr1 * 32 + sc1], A  + (size_t)(m0 + sr1) * D2_ + k0 + sc1);
    async_ld_b128(&sBt[buf][sr0 * 32 + sc0], Wt + (size_t)(n0 + sr0) * D2_ + k0 + sc0);
    async_ld_b128(&sBt[buf][sr1 * 32 + sc1], Wt + (size_t)(n0 + sr1) * D2_ + k0 + sc1);
  };

  v8f acc[2][4] = {};
  stage(0, 0);
  int cur = 0;
  for (int k0 = 0; k0 < D2_; k0 += 32) {
    async_wait_all();
    __syncthreads();
    if (k0 + 32 < D2_) stage(cur ^ 1, k0 + 32);

    v16bf afrag[2], bfrag[4];
    #pragma unroll
    for (int tm = 0; tm < 2; ++tm)
      afrag[tm] = frag_from_contig(&sA[cur][(wm * 32 + tm * 16 + lrow) * 32], hi);
    #pragma unroll
    for (int tn = 0; tn < 4; ++tn)
      bfrag[tn] = frag_from_contig(&sBt[cur][(wn * 64 + tn * 16 + lrow) * 32], hi);
    #pragma unroll
    for (int tm = 0; tm < 2; ++tm)
      #pragma unroll
      for (int tn = 0; tn < 4; ++tn)
        acc[tm][tn] = wmma_bf16f32(afrag[tm], bfrag[tn], acc[tm][tn]);
    cur ^= 1;
  }

  #pragma unroll
  for (int tm = 0; tm < 2; ++tm)
    #pragma unroll
    for (int tn = 0; tn < 4; ++tn)
      #pragma unroll
      for (int r = 0; r < 8; ++r) {
        int row = m0 + wm * 32 + tm * 16 + r + hi * 8;
        int col = n0 + wn * 64 + tn * 16 + lrow;
        Cout[(size_t)row * D_ + col] = acc[tm][tn][r];
      }
}

// ---------------------------------------------------------------------------
// Host launcher
// ---------------------------------------------------------------------------
extern "C" void kernel_launch(void* const* d_in, const int* in_sizes, int n_in,
                              void* d_out, int out_size, void* d_ws, size_t ws_size,
                              hipStream_t stream) {
  (void)in_sizes; (void)n_in; (void)out_size; (void)ws_size;
  const float* X   = (const float*)d_in[0];
  const float* W_q = (const float*)d_in[1];
  const float* W_k = (const float*)d_in[2];
  const float* W_v = (const float*)d_in[3];
  const float* W_o = (const float*)d_in[4];
  const float* lq1 = (const float*)d_in[5];
  const float* lk1 = (const float*)d_in[6];
  const float* lq2 = (const float*)d_in[7];
  const float* lk2 = (const float*)d_in[8];
  const float* rms_scale = (const float*)d_in[9];
  float* out = (float*)d_out;

  char* ws = (char*)d_ws;
  size_t off = 0;
  auto alloc = [&](size_t bytes) -> char* {
    char* p = ws + off;
    off = (off + bytes + 255) & ~(size_t)255;
    return p;
  };
  const size_t nX  = (size_t)B_ * N_ * D_;
  const size_t nW  = (size_t)D_ * D2_;
  const size_t nQ  = (size_t)B_ * H_ * N_ * TDH_;
  const size_t nAO = (size_t)B_ * N_ * D2_;

  bf16_t* Xbf  = (bf16_t*)alloc(nX * 2);
  bf16_t* Wqt  = (bf16_t*)alloc(nW * 2);   // [2048 n][1024 k]
  bf16_t* Wkt  = (bf16_t*)alloc(nW * 2);
  bf16_t* Wvt  = (bf16_t*)alloc(nW * 2);
  bf16_t* Wot  = (bf16_t*)alloc(nW * 2);   // [1024 n][2048 k]
  bf16_t* Qb   = (bf16_t*)alloc(nQ * 2);   // [B,H,N,256]
  bf16_t* Kb   = (bf16_t*)alloc(nQ * 2);   // [B,H,N,256]
  bf16_t* Vtb  = (bf16_t*)alloc(nQ * 2);   // [B,H,256,N]
  bf16_t* AOb  = (bf16_t*)alloc(nAO * 2);  // [B,N,2048]
  float*  lam  = (float*)alloc(H_ * sizeof(float));

  // 1) convert (+transpose weights) fp32 -> bf16
  cvt_f32_bf16<<<(int)((nX + 255) / 256), 256, 0, stream>>>(X, Xbf, (int)nX);
  cvt_transpose_bf16<<<dim3(D2_ / 32, D_ / 32), 256, 0, stream>>>(W_q, Wqt, D_, D2_);
  cvt_transpose_bf16<<<dim3(D2_ / 32, D_ / 32), 256, 0, stream>>>(W_k, Wkt, D_, D2_);
  cvt_transpose_bf16<<<dim3(D2_ / 32, D_ / 32), 256, 0, stream>>>(W_v, Wvt, D_, D2_);
  cvt_transpose_bf16<<<dim3(D_ / 32, D2_ / 32), 256, 0, stream>>>(W_o, Wot, D2_, D_);

  // 2) lambda per head
  lambda_kernel<<<1, 256, 0, stream>>>(lq1, lk1, lq2, lk2, lam);

  // 3) fused QKV projection (z selects q/k/v; V written transposed)
  qkv_proj_kernel<<<dim3(D2_ / 128, (B_ * N_) / 128, 3), 256, 0, stream>>>(
      Xbf, Wqt, Wkt, Wvt, Qb, Kb, Vtb);

  // 4) differential flash attention + RMS norm
  hipFuncSetAttribute((const void*)diff_attn_kernel,
                      hipFuncAttributeMaxDynamicSharedMemorySize, SMEM_ATTN);
  diff_attn_kernel<<<dim3(N_ / 64, B_ * H_), 256, SMEM_ATTN, stream>>>(
      Qb, Kb, Vtb, lam, rms_scale, AOb);

  // 5) output projection -> fp32
  out_proj_kernel<<<dim3(D_ / 128, (B_ * N_) / 128), 256, 0, stream>>>(AOb, Wot, out);
}
